// DeformConv2D_81853486727283
// MI455X (gfx1250) — matrix operand order repaired and checked by reading, not compiled, not run on hardware
//
#include <hip/hip_runtime.h>
#include <hip/hip_bf16.h>
#include <stdint.h>

// Problem constants (match reference)
#define B_     8
#define CIN_   128
#define COUT_  128
#define H_     64
#define W_     64
#define HW_    4096          // H_*W_
#define N_     9             // KS*KS
#define K_     1152          // N_*CIN_
#define KBLK_  36            // K_/32
// Workspace layout (bytes, 256-aligned partitions; total ~28.9 MB)
#define WS_OFF    0            // off:   8*18*4096*4      = 2,359,296
#define WS_XT     2359296      // x_t:   8*4096*128*4     = 16,777,216
#define WS_IDX    19136512     // idx4:  294912*4*4       = 4,718,592
#define WS_WGT    23855104     // wgt4:  294912*4*4       = 4,718,592
#define WS_WFRAG  28573696     // wfrag: 36*8*32*16*2     = 294,912

typedef __attribute__((ext_vector_type(16))) _Float16 v16h_t;
typedef __attribute__((ext_vector_type(8)))  float    v8f_t;

// ---------------------------------------------------------------------------
// Kernel 1: offset conv  (b,18,h,w) = conv3x3(x, w_off, pad=1) + b_off
// One block = one batch x 256 pixels; all 18 out-channels per thread.
// ---------------------------------------------------------------------------
__global__ void __launch_bounds__(256) off_conv_kernel(
    const float* __restrict__ x, const float* __restrict__ w_off,
    const float* __restrict__ b_off, float* __restrict__ off) {
  __shared__ float wl[18 * 1152];          // full w_off: 82.9 KB (of 320 KB)
  int b   = blockIdx.x >> 4;
  int pix = ((blockIdx.x & 15) << 8) + threadIdx.x;
  for (int i = threadIdx.x; i < 18 * 1152; i += 256) wl[i] = w_off[i];
  __syncthreads();
  int h = pix >> 6, w = pix & 63;
  float acc[18];
#pragma unroll
  for (int oc = 0; oc < 18; ++oc) acc[oc] = 0.f;
  const float* xb = x + b * CIN_ * HW_;
  for (int c = 0; c < CIN_; ++c) {
    const float* xc = xb + c * HW_;
    float t[9];
#pragma unroll
    for (int kh = 0; kh < 3; ++kh) {
      int hh = h + kh - 1;
#pragma unroll
      for (int kw = 0; kw < 3; ++kw) {
        int ww = w + kw - 1;
        bool ok = ((unsigned)hh < 64u) && ((unsigned)ww < 64u);
        t[kh * 3 + kw] = ok ? xc[hh * 64 + ww] : 0.f;
      }
    }
    const float* wc = wl + c * 9;
#pragma unroll
    for (int oc = 0; oc < 18; ++oc) {
      const float* wo = wc + oc * 1152;
#pragma unroll
      for (int k = 0; k < 9; ++k) acc[oc] += t[k] * wo[k];
    }
  }
#pragma unroll
  for (int oc = 0; oc < 18; ++oc)
    off[(b * 18 + oc) * HW_ + pix] = acc[oc] + b_off[oc];
}

// ---------------------------------------------------------------------------
// Kernel 2: per-(b,pixel,n) bilinear corner indices + weights (with the
// reference's overlap correction folded into the weights).
// ---------------------------------------------------------------------------
__global__ void __launch_bounds__(256) prep_kernel(
    const float* __restrict__ off, int* __restrict__ idx4,
    float* __restrict__ wgt4) {
  int t = blockIdx.x * 256 + threadIdx.x;        // 0 .. 294911
  if (t >= B_ * HW_ * N_) return;
  int n = t % 9;
  int pix = (t / 9) & (HW_ - 1);
  int b = t / (9 * HW_);
  int h = pix >> 6, w = pix & 63;
  float offx = off[(b * 18 + n)     * HW_ + pix];   // x = row direction
  float offy = off[(b * 18 + 9 + n) * HW_ + pix];   // y = col direction
  float px = (float)(h + (n / 3) - 1) + offx;
  float py = (float)(w + (n % 3) - 1) + offy;
  px = fminf(fmaxf(px, 0.f), 63.f);
  py = fminf(fmaxf(py, 0.f), 63.f);
  float fx = floorf(px), cx = ceilf(px);
  float fy = floorf(py), cy = ceilf(py);
  float gx0 = 1.f - (px - fx), gx1 = 1.f - (cx - px);
  float gy0 = 1.f - (py - fy), gy1 = 1.f - (cy - py);
  float inv = 1.f / (((fx == cx) ? 2.f : 1.f) * ((fy == cy) ? 2.f : 1.f));
  int ifx = (int)fx, icx = (int)cx, ify = (int)fy, icy = (int)cy;
  int4   qi = make_int4(ifx * 64 + ify, icx * 64 + icy,
                        ifx * 64 + icy, icx * 64 + ify);   // lt, rb, lb, rt
  float4 qw = make_float4(gx0 * gy0 * inv, gx1 * gy1 * inv,
                          gx0 * gy1 * inv, gx1 * gy0 * inv);
  ((int4*)idx4)[t]   = qi;
  ((float4*)wgt4)[t] = qw;
}

// ---------------------------------------------------------------------------
// Kernel 3: transpose x (b,c,hw) -> x_t (b,hw,c) so gathers read contiguous
// 128B channel chunks. 64-pixel x 128-channel LDS tiles (129 pad = no bank
// conflicts).
// ---------------------------------------------------------------------------
__global__ void __launch_bounds__(256) transpose_kernel(
    const float* __restrict__ x, float* __restrict__ x_t) {
  __shared__ float tile[64 * 129];
  int b  = blockIdx.x >> 6;
  int p0 = (blockIdx.x & 63) * 64;
  const float* xb = x + b * CIN_ * HW_;
  for (int i = threadIdx.x; i < 64 * 128; i += 256) {
    int c = i >> 6, p = i & 63;                    // coalesced over p
    tile[p * 129 + c] = xb[c * HW_ + p0 + p];
  }
  __syncthreads();
  float* ob = x_t + (b * HW_ + p0) * CIN_;
  for (int i = threadIdx.x; i < 64 * 128; i += 256) {
    int p = i >> 7, c = i & 127;                   // contiguous store rows
    ob[i] = tile[p * 129 + c];
  }
}

// ---------------------------------------------------------------------------
// Kernel 4: pre-swizzle w_ker into per-lane B fragments (f16).
// B 32x16 layout: column N = lane%16, K = 16*(e>=8) + 8*(lane/16) + (e&7)  (e = 0..15).
// Stored as wfrag[kblk][wave][lane][e] so the GEMM reads 32 contiguous bytes
// per lane (fully coalesced).  K order: k = n*128 + c.
// ---------------------------------------------------------------------------
__global__ void __launch_bounds__(256) wfrag_kernel(
    const float* __restrict__ w_ker, _Float16* __restrict__ wfrag) {
  int t = blockIdx.x * 256 + threadIdx.x;          // 0 .. 147455
  if (t >= KBLK_ * 8 * 32 * 16) return;
  int e    = t & 15;
  int lane = (t >> 4) & 31;
  int wv   = (t >> 9) & 7;
  int kblk = t >> 12;
  int hf   = lane >> 4;
  int ncol = lane & 15;
  int k = kblk * 32 + ((e >> 3) << 4) + hf * 8 + (e & 7); // global K index
  int n = k >> 7;                                  // k = n*128 + c
  int c = k & 127;
  int cout = wv * 16 + ncol;
  wfrag[t] = (_Float16)w_ker[(cout * CIN_ + c) * 9 + n];
}

// ---------------------------------------------------------------------------
// Kernel 5: fused bilinear-sample + GEMM via v_wmma_f32_16x16x32_f16.
// Block = 16-pixel M-tile, 8 waves = 8 x 16-cout N-tiles (all 128 couts).
// Outer loop over the 9 taps n: load the 4 corner indices/weights ONCE per
// tap (not per 32-K step), build the full 16x128 f16 A-slab for that tap in
// LDS (row stride 136 halfs = 272B: every 16B fragment chunk stays 16B
// aligned; bank = 4*m mod 64 -> conflict-free), then run 4 unrolled WMMA
// K-steps against it.  2 barriers per tap (18 total) instead of 2 per K-step.
// A 16x32 f16 fragment layout (per 32-K window at half-offset c0):
//   M = lane%16, element e -> K = c0 + 16*(e>=8) + 8*half + (e&7)
//   => elements 0..7  = halfs [c0+8*half     .. +7]  (one aligned 16B chunk)
//      elements 8..15 = halfs [c0+16+8*half  .. +7]  (one aligned 16B chunk)
// ---------------------------------------------------------------------------
__global__ void __launch_bounds__(256) deform_gemm_kernel(
    const float* __restrict__ x_t, const int* __restrict__ idx4,
    const float* __restrict__ wgt4, const _Float16* __restrict__ wfrag,
    const float* __restrict__ b_ker, float* __restrict__ out) {
  __shared__ _Float16 A_lds[16 * 136];             // 16 rows, 136-half stride
  int mtile = blockIdx.x;                          // 0..2047
  int b    = mtile >> 8;                           // 256 M-tiles per batch
  int pix0 = (mtile & 255) << 4;
  int tid  = threadIdx.x;
  int lane = tid & 31;
  int wv   = tid >> 5;
  int hf   = lane >> 4;
  int mr   = lane & 15;
  int m_a  = tid >> 5;                             // A-build row (0..7)
  int cc   = tid & 31;                             // A-build channel-in-block
  const float* xb = x_t + b * HW_ * CIN_;
  int base_mn = (b * HW_ + pix0) * 9;
  v8f_t acc = {};
  for (int n = 0; n < N_; ++n) {
    // Corner indices / weights for this tap: rows m_a and m_a+8, loaded once.
    int item0 = base_mn + m_a * 9 + n;
    int item1 = base_mn + (m_a + 8) * 9 + n;
    int4   qi0 = *(const int4*)(idx4 + item0 * 4);
    float4 qw0 = *(const float4*)(wgt4 + item0 * 4);
    int4   qi1 = *(const int4*)(idx4 + item1 * 4);
    float4 qw1 = *(const float4*)(wgt4 + item1 * 4);
    __syncthreads();                               // LDS WAR vs prev tap
#pragma unroll
    for (int cb = 0; cb < 4; ++cb) {               // build full 16x128 slab
      int c = cb * 32 + cc;
      float v0 = qw0.x * xb[qi0.x * CIN_ + c] + qw0.y * xb[qi0.y * CIN_ + c]
               + qw0.z * xb[qi0.z * CIN_ + c] + qw0.w * xb[qi0.w * CIN_ + c];
      float v1 = qw1.x * xb[qi1.x * CIN_ + c] + qw1.y * xb[qi1.y * CIN_ + c]
               + qw1.z * xb[qi1.z * CIN_ + c] + qw1.w * xb[qi1.w * CIN_ + c];
      A_lds[m_a * 136 + c]       = (_Float16)v0;
      A_lds[(m_a + 8) * 136 + c] = (_Float16)v1;
    }
    __syncthreads();
#pragma unroll
    for (int cb = 0; cb < 4; ++cb) {               // 4 WMMA K-steps per tap
      int kblk = n * 4 + cb;
      int c0 = cb * 32;
      union { uint4 u[2]; v16h_t v; } ua, ub;
      ua.u[0] = *(const uint4*)(A_lds + mr * 136 + c0 +      hf * 8);
      ua.u[1] = *(const uint4*)(A_lds + mr * 136 + c0 + 16 + hf * 8);
      const uint4* bp =
          (const uint4*)(wfrag + (((kblk * 8 + wv) * 32 + lane) << 4));
      ub.u[0] = bp[0];
      ub.u[1] = bp[1];
      acc = __builtin_amdgcn_wmma_f32_16x16x32_f16(
          /*neg_a=*/false, ua.v, /*neg_b=*/false, ub.v,
          /*c_mod=*/(short)0, acc, /*reuse_a=*/false, /*reuse_b=*/false);
    }
  }
  // D layout: M = vgpr + 8*half, N = lane%16 -> 8 consecutive pixels per lane
  int cout = wv * 16 + (lane & 15);
  float bias = b_ker[cout];
  float* ob = out + (b * COUT_ + cout) * HW_ + pix0 + hf * 8;
#pragma unroll
  for (int v = 0; v < 8; ++v) ob[v] = acc[v] + bias;
}

// ---------------------------------------------------------------------------
extern "C" void kernel_launch(void* const* d_in, const int* in_sizes, int n_in,
                              void* d_out, int out_size, void* d_ws, size_t ws_size,
                              hipStream_t stream) {
  (void)in_sizes; (void)n_in; (void)out_size; (void)ws_size;
  const float* x     = (const float*)d_in[0];
  const float* w_off = (const float*)d_in[1];
  const float* b_off = (const float*)d_in[2];
  const float* w_ker = (const float*)d_in[3];
  const float* b_ker = (const float*)d_in[4];
  float* out = (float*)d_out;

  char* ws = (char*)d_ws;
  float*    off   = (float*)(ws + WS_OFF);
  float*    x_t   = (float*)(ws + WS_XT);
  int*      idx4  = (int*)  (ws + WS_IDX);
  float*    wgt4  = (float*)(ws + WS_WGT);
  _Float16* wfrag = (_Float16*)(ws + WS_WFRAG);

  off_conv_kernel  <<<dim3(128),  dim3(256), 0, stream>>>(x, w_off, b_off, off);
  prep_kernel      <<<dim3(1152), dim3(256), 0, stream>>>(off, idx4, wgt4);
  transpose_kernel <<<dim3(512),  dim3(256), 0, stream>>>(x, x_t);
  wfrag_kernel     <<<dim3(576),  dim3(256), 0, stream>>>(w_ker, wfrag);
  deform_gemm_kernel<<<dim3(2048), dim3(256), 0, stream>>>(x_t, idx4, wgt4,
                                                           wfrag, b_ker, out);
}
